// myGNN_17454747091496
// MI455X (gfx1250) — matrix operand order repaired and checked by reading, not compiled, not run on hardware
//
#include <hip/hip_runtime.h>
#include <hip/hip_bf16.h>

// ---------------------------------------------------------------------------
// Types for CDNA5 WMMA
// ---------------------------------------------------------------------------
typedef __attribute__((ext_vector_type(16))) __bf16 v16bf;
typedef __attribute__((ext_vector_type(2)))  __bf16 v2bf;
typedef __attribute__((ext_vector_type(8)))  float  v8f;

#define NEG_SLOPE 0.01f

// float -> bf16 (round to nearest even), raw u16 bits (one-shot convert kernel)
static __device__ __forceinline__ unsigned short f2bf_bits(float f) {
    unsigned u = __builtin_bit_cast(unsigned, f);
    unsigned r = u + 0x7FFFu + ((u >> 16) & 1u);
    return (unsigned short)(r >> 16);
}

static __device__ __forceinline__ v8f zero8() {
    v8f z = {0.f, 0.f, 0.f, 0.f, 0.f, 0.f, 0.f, 0.f};
    return z;
}

static __device__ __forceinline__ float lrelu(float v) {
    return (v > 0.f) ? v : NEG_SLOPE * v;
}

// ---------------------------------------------------------------------------
// WMMA fragment loads from LDS (row-major bf16, ldk multiple of 8)
// A (16x32 frag): lanes 0-15 hold row=lane, K 0-7 & 16-23;
//                 lanes 16-31 hold row=lane-16, K 8-15 & 24-31.
// B (32x16 frag): lane&15 = output column (= weight row), lanes<16 K 0-7
//                 & 16-23, lanes>=16 K 8-15 & 24-31.
// ---------------------------------------------------------------------------
static __device__ __forceinline__ v16bf load_frag_a(const __bf16* tile,
                                                    int ldk, int lane) {
    int row = lane & 15;
    int k0  = (lane < 16) ? 0 : 8;
    const __bf16* p = tile + row * ldk + k0;
    union { v16bf v; uint4 q[2]; } u;
    u.q[0] = *(const uint4*)(p);
    u.q[1] = *(const uint4*)(p + 16);
    return u.v;
}
static __device__ __forceinline__ v16bf load_frag_b(const __bf16* w, int ldk,
                                                    int colbase, int kbase,
                                                    int lane) {
    int row = colbase + (lane & 15);
    int k0  = kbase + ((lane < 16) ? 0 : 8);
    const __bf16* p = w + row * ldk + k0;
    union { v16bf v; uint4 q[2]; } u;
    u.q[0] = *(const uint4*)(p);
    u.q[1] = *(const uint4*)(p + 16);
    return u.v;
}

// One dense layer on a 16-row tile: out = lrelu(in @ W^T + b), bf16 in LDS.
// A fragments hoisted into VGPRs and reused across all output col-tiles.
// Epilogue packs row-pairs with one v_cvt_pk_bf16_f32; the high-half store
// lowers to ds_store_b16_d16_hi.
template <int KIN, int NOUT>
static __device__ __forceinline__ void mlp_layer(const __bf16* in, __bf16* out,
                                                 const __bf16* w,
                                                 const float* biasLds,
                                                 int lane) {
    constexpr int NKT = KIN / 32;
    v16bf af[NKT];
#pragma unroll
    for (int kt = 0; kt < NKT; ++kt)
        af[kt] = load_frag_a(in + kt * 32, KIN, lane);
#pragma unroll
    for (int ct = 0; ct < NOUT / 16; ++ct) {
        v8f acc = zero8();
#pragma unroll
        for (int kt = 0; kt < NKT; ++kt) {
            v16bf b = load_frag_b(w, KIN, ct * 16, kt * 32, lane);
            acc = __builtin_amdgcn_wmma_f32_16x16x32_bf16(
                false, af[kt], false, b, (short)0, acc, false, false);
        }
        int   col = ct * 16 + (lane & 15);
        float bv  = biasLds[col];
#pragma unroll
        for (int r = 0; r < 8; r += 2) {
            int   row0 = ((lane < 16) ? r : (r + 8));
            float v0   = lrelu(acc[r] + bv);
            float v1   = lrelu(acc[r + 1] + bv);
            v2bf  p    = {(__bf16)v0, (__bf16)v1};  // one v_cvt_pk_bf16_f32
            out[row0 * NOUT + col]       = p[0];
            out[(row0 + 1) * NOUT + col] = p[1];    // d16_hi store
        }
    }
}

// ---------------------------------------------------------------------------
// Kernel 0: convert weights fp32 -> bf16, contiguous in d_ws:
// [w1 256*64][w2 128*256][w3 64*128][w_self 64*64][w_neigh 64*64][w4 64]
// ---------------------------------------------------------------------------
__global__ void gnn_convert_weights(const float* __restrict__ w1,
                                    const float* __restrict__ w2,
                                    const float* __restrict__ w3,
                                    const float* __restrict__ wself,
                                    const float* __restrict__ wneigh,
                                    const float* __restrict__ w4,
                                    unsigned short* __restrict__ out) {
    const int S1 = 256 * 64, S2 = 128 * 256, S3 = 64 * 128, S4 = 64 * 64;
    const int T1 = S1, T2 = T1 + S2, T3 = T2 + S3, T4 = T3 + S4, T5 = T4 + S4;
    int i = blockIdx.x * blockDim.x + threadIdx.x;
    if (i >= T5 + 64) return;
    float v;
    if (i < T1)      v = w1[i];
    else if (i < T2) v = w2[i - T1];
    else if (i < T3) v = w3[i - T2];
    else if (i < T4) v = wself[i - T3];
    else if (i < T5) v = wneigh[i - T4];
    else             v = w4[i - T5];
    out[i] = f2bf_bits(v);
}

__global__ void gnn_zero(float* __restrict__ p, int n) {
    int i = blockIdx.x * blockDim.x + threadIdx.x;
    for (; i < n; i += gridDim.x * blockDim.x) p[i] = 0.f;
}

// ---------------------------------------------------------------------------
// Kernel 1: per-edge MLP (WMMA). One wave handles one 16-edge tile at a time.
// LDS: [b1 256f][b2 128f][b3 64f][w4 64bf][W1|W2|W3 bf16 112K][8 x 12KB act]
// ---------------------------------------------------------------------------
#define EDGE_WAVES 8
#define EDGE_ACT 6144  // bf16 elems per wave (12KB)

__global__ void gnn_edge_mlp(const float* __restrict__ x,
                             const int* __restrict__ src,
                             const int* __restrict__ dst,
                             const unsigned short* __restrict__ wall,
                             const float* __restrict__ b1,
                             const float* __restrict__ b2,
                             const float* __restrict__ b3,
                             const float* __restrict__ b4,
                             float* __restrict__ e_out, int E) {
    extern __shared__ char smem_raw[];
    float*  sB1  = (float*)smem_raw;            // 256 @ 0
    float*  sB2  = sB1 + 256;                   // 128 @ 1024B
    float*  sB3  = sB2 + 128;                   // 64  @ 1536B
    __bf16* sW4b = (__bf16*)(smem_raw + 1792);  // 64  @ 1792B (128B)
    __bf16* sW1  = (__bf16*)(smem_raw + 2048);  // 16384
    __bf16* sW2  = sW1 + 16384;                 // 32768
    __bf16* sW3  = sW1 + 49152;                 // 8192
    __bf16* act  = sW1 + 57344;                 // per-wave activation

    // Cooperative staging: 57344 bf16 = 7168 uint4, fp32 biases, bf16 w4
    {
        const uint4* s = (const uint4*)wall;
        uint4*       d = (uint4*)sW1;
        for (int i = threadIdx.x; i < 7168; i += blockDim.x) d[i] = s[i];
        for (int i = threadIdx.x; i < 256; i += blockDim.x) sB1[i] = b1[i];
        for (int i = threadIdx.x; i < 128; i += blockDim.x) sB2[i] = b2[i];
        if (threadIdx.x < 64) sB3[threadIdx.x] = b3[threadIdx.x];
        if (threadIdx.x < 8)
            ((uint4*)sW4b)[threadIdx.x] =
                ((const uint4*)(wall + 65536))[threadIdx.x];
    }
    __syncthreads();

    const int lane = threadIdx.x & 31;
    const int wave = threadIdx.x >> 5;
    __bf16* my    = act + wave * EDGE_ACT;
    __bf16* bufD  = my;          // 16x64  @ [0,1024)
    __bf16* bufH1 = my + 2048;   // 16x256 @ [2048,6144)
    __bf16* bufH2 = my;          // 16x128 @ [0,2048)
    __bf16* bufH3 = my + 2048;   // 16x64  @ [2048,3072)

    const float b4v = b4[0];
    int gw      = blockIdx.x * EDGE_WAVES + wave;
    int gstride = gridDim.x * EDGE_WAVES;
    int ntiles  = (E + 15) >> 4;

    for (int t = gw; t < ntiles; t += gstride) {
        int ebase = t << 4;
        // prefetch next tile's edge indices (global_prefetch_b8)
        int tn = t + gstride;
        if (tn < ntiles) {
            __builtin_prefetch(src + (tn << 4), 0, 0);
            __builtin_prefetch(dst + (tn << 4), 0, 0);
        }
        // gather d = x[src]-x[dst]: lane -> edge ebase+(lane>>1), half (lane&1)*32
        {
            int eidx = ebase + (lane >> 1);
            int ec   = (eidx < E) ? eidx : (E - 1);
            int half = (lane & 1) * 32;
            const float* xs   = x + (size_t)src[ec] * 64 + half;
            const float* xd   = x + (size_t)dst[ec] * 64 + half;
            __bf16*      drow = bufD + (lane >> 1) * 64 + half;
#pragma unroll
            for (int c = 0; c < 8; ++c) {
                float4 a = *(const float4*)(xs + c * 4);
                float4 b = *(const float4*)(xd + c * 4);
                drow[c * 4 + 0] = (__bf16)(a.x - b.x);
                drow[c * 4 + 1] = (__bf16)(a.y - b.y);
                drow[c * 4 + 2] = (__bf16)(a.z - b.z);
                drow[c * 4 + 3] = (__bf16)(a.w - b.w);
            }
        }
        mlp_layer<64, 256>(bufD, bufH1, sW1, sB1, lane);   // 32 WMMA
        mlp_layer<256, 128>(bufH1, bufH2, sW2, sB2, lane); // 64 WMMA
        mlp_layer<128, 64>(bufH2, bufH3, sW3, sB3, lane);  // 16 WMMA

        // layer 4: dot(64) + sigmoid. Two lanes per edge: lane&15 = edge,
        // lane>>4 picks 32-element half; packed-bf16 u32 unpacking.
        {
            const unsigned* hrow =
                (const unsigned*)(bufH3 + (lane & 15) * 64) + (lane >> 4) * 16;
            const unsigned* wrow = (const unsigned*)sW4b + (lane >> 4) * 16;
            float s = 0.f;
#pragma unroll
            for (int k = 0; k < 16; ++k) {
                unsigned hu = hrow[k], wu = wrow[k];
                float hl = __builtin_bit_cast(float, hu << 16);
                float hh = __builtin_bit_cast(float, hu & 0xffff0000u);
                float wl = __builtin_bit_cast(float, wu << 16);
                float wh = __builtin_bit_cast(float, wu & 0xffff0000u);
                s = fmaf(hl, wl, s);
                s = fmaf(hh, wh, s);
            }
            s += __shfl_xor(s, 16, 32);
            s += b4v;
            float ev = 1.f / (1.f + __expf(-s));
            int   eo = ebase + (lane & 15);
            if (lane < 16 && eo < E) e_out[eo] = ev;
        }
    }
}

// ---------------------------------------------------------------------------
// Kernel 2: weighted mean aggregation via f32 atomics (sacc += e*x[src], deg++)
// ---------------------------------------------------------------------------
__global__ void gnn_aggregate(const float* __restrict__ x,
                              const int* __restrict__ src,
                              const int* __restrict__ dst,
                              const float* __restrict__ e,
                              float* __restrict__ sacc,
                              float* __restrict__ deg, int E) {
    int t    = blockIdx.x * blockDim.x + threadIdx.x;
    int edge = t >> 1;
    if (edge >= E) return;
    int   half = (t & 1) * 32;
    float ev   = e[edge];
    const float* xs  = x + (size_t)src[edge] * 64 + half;
    float*       out = sacc + (size_t)dst[edge] * 64 + half;
#pragma unroll
    for (int c = 0; c < 32; ++c)
        __hip_atomic_fetch_add(out + c, ev * xs[c], __ATOMIC_RELAXED,
                               __HIP_MEMORY_SCOPE_AGENT);
    if ((t & 1) == 0)
        __hip_atomic_fetch_add(deg + dst[edge], 1.0f, __ATOMIC_RELAXED,
                               __HIP_MEMORY_SCOPE_AGENT);
}

// ---------------------------------------------------------------------------
// Kernel 3: node update. A = lrelu(x@Wself^T + neigh@Wneigh^T + b); L2-normalize.
// In-place over the accumulator region (each 16-node tile owned by one wave).
// ---------------------------------------------------------------------------
#define NODE_WAVES 8

__global__ void gnn_node_update(const float* __restrict__ x,
                                const float* __restrict__ deg,
                                const unsigned short* __restrict__ wsb,
                                const float* __restrict__ bias,
                                float* __restrict__ A /* also sacc in */,
                                int N) {
    extern __shared__ char smem_raw[];
    float*  sBias = (float*)smem_raw;            // 64 @ 0 (256 B)
    __bf16* sWS   = (__bf16*)(smem_raw + 256);   // 64*64
    __bf16* sWN   = sWS + 4096;                  // 64*64
    __bf16* act   = sWN + 4096;                  // per wave 2048 bf16
    {
        const uint4* s = (const uint4*)wsb;  // w_self,w_neigh contiguous 8192 u16
        uint4*       d = (uint4*)sWS;
        for (int i = threadIdx.x; i < 1024; i += blockDim.x) d[i] = s[i];
        if (threadIdx.x < 64) sBias[threadIdx.x] = bias[threadIdx.x];
    }
    __syncthreads();

    const int lane = threadIdx.x & 31;
    const int wave = threadIdx.x >> 5;
    __bf16* xb = act + wave * 2048;  // 16x64
    __bf16* nb = xb + 1024;          // 16x64

    int gw      = blockIdx.x * NODE_WAVES + wave;
    int gstride = gridDim.x * NODE_WAVES;
    int ntiles  = (N + 15) >> 4;

    for (int t = gw; t < ntiles; t += gstride) {
        int nb0 = t << 4;
        // stage x and neigh = sacc/max(deg,1) tiles as bf16
        {
            int node = nb0 + (lane >> 1);
            int nc   = (node < N) ? node : (N - 1);
            int half = (lane & 1) * 32;
            float inv = 1.f / fmaxf(deg[nc], 1.f);
            const float* xr   = x + (size_t)nc * 64 + half;
            const float* sr   = A + (size_t)nc * 64 + half;
            __bf16*      xrow = xb + (lane >> 1) * 64 + half;
            __bf16*      nrow = nb + (lane >> 1) * 64 + half;
#pragma unroll
            for (int c = 0; c < 32; ++c) {
                xrow[c] = (__bf16)xr[c];
                nrow[c] = (__bf16)(sr[c] * inv);
            }
        }
        // hoisted A fragments (K=64: 2 frags each for x and neigh)
        v16bf ax[2], an[2];
#pragma unroll
        for (int kt = 0; kt < 2; ++kt) {
            ax[kt] = load_frag_a(xb + kt * 32, 64, lane);
            an[kt] = load_frag_a(nb + kt * 32, 64, lane);
        }
        float vout[4][8];
#pragma unroll
        for (int ct = 0; ct < 4; ++ct) {
            v8f acc = zero8();
#pragma unroll
            for (int kt = 0; kt < 2; ++kt) {
                v16bf b = load_frag_b(sWS, 64, ct * 16, kt * 32, lane);
                acc = __builtin_amdgcn_wmma_f32_16x16x32_bf16(
                    false, ax[kt], false, b, (short)0, acc, false, false);
            }
#pragma unroll
            for (int kt = 0; kt < 2; ++kt) {
                v16bf b = load_frag_b(sWN, 64, ct * 16, kt * 32, lane);
                acc = __builtin_amdgcn_wmma_f32_16x16x32_bf16(
                    false, an[kt], false, b, (short)0, acc, false, false);
            }
            float bv = sBias[ct * 16 + (lane & 15)];
#pragma unroll
            for (int r = 0; r < 8; ++r) vout[ct][r] = lrelu(acc[r] + bv);
        }
        // row-wise sum of squares: 4 col-tiles, then across 16 lanes per half
        float ss[8];
#pragma unroll
        for (int r = 0; r < 8; ++r) {
            ss[r] = vout[0][r] * vout[0][r] + vout[1][r] * vout[1][r] +
                    vout[2][r] * vout[2][r] + vout[3][r] * vout[3][r];
        }
#pragma unroll
        for (int m = 1; m < 16; m <<= 1) {
#pragma unroll
            for (int r = 0; r < 8; ++r) ss[r] += __shfl_xor(ss[r], m, 32);
        }
#pragma unroll
        for (int r = 0; r < 8; ++r) {
            int   row  = nb0 + ((lane < 16) ? r : (r + 8));
            float invn = 1.f / fmaxf(sqrtf(ss[r]), 1e-12f);
            if (row < N) {
#pragma unroll
                for (int ct = 0; ct < 4; ++ct) {
                    int col = ct * 16 + (lane & 15);
                    A[(size_t)row * 64 + col] = vout[ct][r] * invn;
                }
            }
        }
    }
}

// ---------------------------------------------------------------------------
// Launch
// ---------------------------------------------------------------------------
extern "C" void kernel_launch(void* const* d_in, const int* in_sizes, int n_in,
                              void* d_out, int out_size, void* d_ws,
                              size_t ws_size, hipStream_t stream) {
    const float* x      = (const float*)d_in[0];
    const int*   src    = (const int*)d_in[1];
    const int*   dst    = (const int*)d_in[2];
    const float* w1     = (const float*)d_in[3];
    const float* b1     = (const float*)d_in[4];
    const float* w2     = (const float*)d_in[5];
    const float* b2     = (const float*)d_in[6];
    const float* w3     = (const float*)d_in[7];
    const float* b3     = (const float*)d_in[8];
    const float* w4     = (const float*)d_in[9];
    const float* b4     = (const float*)d_in[10];
    const float* wself  = (const float*)d_in[11];
    const float* wneigh = (const float*)d_in[12];
    const float* bias   = (const float*)d_in[13];

    const int N = in_sizes[0] / 64;
    const int E = in_sizes[1];

    float* A_out = (float*)d_out;            // [N,64] (also sacc accumulator)
    float* e_out = A_out + (size_t)N * 64;   // [E]

    unsigned short* wall = (unsigned short*)d_ws;           // 65600 u16
    unsigned short* wsb  = wall + 57344;                    // w_self/w_neigh bf16
    float*          deg  = (float*)((char*)d_ws + 131584);  // [N]

    // 0: weight conversion (65600 elements)
    gnn_convert_weights<<<(65600 + 255) / 256, 256, 0, stream>>>(
        w1, w2, w3, wself, wneigh, w4, wall);
    // 0b: zero accumulators
    gnn_zero<<<1024, 256, 0, stream>>>(A_out, N * 64);
    gnn_zero<<<256, 256, 0, stream>>>(deg, N);

    // 1: edge MLP (WMMA)
    const int edge_smem = 2048 + 57344 * 2 + EDGE_WAVES * EDGE_ACT * 2; // 215040 B
    gnn_edge_mlp<<<1024, EDGE_WAVES * 32, edge_smem, stream>>>(
        x, src, dst, wall, b1, b2, b3, b4, e_out, E);

    // 2: aggregation (2 threads per edge)
    gnn_aggregate<<<(2 * E + 255) / 256, 256, 0, stream>>>(x, src, dst, e_out,
                                                           A_out, deg, E);

    // 3: node update (WMMA) — in place over A_out
    const int node_smem = 256 + 8192 * 2 + NODE_WAVES * 2048 * 2;  // 49408 B
    gnn_node_update<<<512, NODE_WAVES * 32, node_smem, stream>>>(
        x, deg, wsb, bias, A_out, N);
    (void)n_in; (void)out_size; (void)ws_size;
}
